// GraphModel_65017214927401
// MI455X (gfx1250) — matrix operand order repaired and checked by reading, not compiled, not run on hardware
//
#include <hip/hip_runtime.h>
#include <hip/hip_bf16.h>
#include <math.h>

// ---------------------------------------------------------------------------
// GATv2 x2 + global attention pooling, MI455X (gfx1250, wave32, WMMA)
// ---------------------------------------------------------------------------

typedef __attribute__((ext_vector_type(16))) __bf16 v16bf;
typedef __attribute__((ext_vector_type(8)))  float  v8f;

#define N_NODES   65536
#define NPG       32        // nodes per graph
#define N_GRAPHS  2048
#define EPG       256       // edges per graph
#define HEADS     4
#define DIM       64        // hidden dim per head (both layers)
#define HD        (HEADS * DIM)   // 256

__device__ __forceinline__ __bf16 f2bf(float f) {
  unsigned u = __builtin_bit_cast(unsigned, f);
  u = (u + 0x7FFFu + ((u >> 16) & 1u)) >> 16;   // round-to-nearest-even
  unsigned short s = (unsigned short)u;
  return __builtin_bit_cast(__bf16, s);
}

// ---------------------------------------------------------------------------
// One-shot weight pre-pack: W[K,256] fp32 -> bf16 in WMMA B-fragment order.
// Packed element index = ((nt*KB + kb)*32 + lane)*16 + i, where the fragment
// for (n-tile nt, k-block kb) on `lane` is 16 contiguous bf16 (32 bytes):
//   n = nt*16 + (lane&15),  k = kb*32 + (lane>>4)*16 + i
// GEMM then reads each B fragment as two coalesced b128 loads, no VALU conv.
// ---------------------------------------------------------------------------
template <int K>
__global__ __launch_bounds__(256) void pack_w_kernel(
    const float* __restrict__ W, __bf16* __restrict__ P) {
  constexpr int KB    = K / 32;
  constexpr int LOGKB = (KB == 4) ? 2 : 1;
  const int idx  = blockIdx.x * 256 + threadIdx.x;      // < K*256
  const int i    = idx & 15;
  const int lane = (idx >> 4) & 31;
  const int kb   = (idx >> 9) & (KB - 1);
  const int nt   = idx >> (9 + LOGKB);
  const int n = nt * 16 + (lane & 15);
  const int k = kb * 32 + (lane >> 4) * 16 + i;
  P[idx] = f2bf(W[(size_t)k * HD + n]);
}

// ---------------------------------------------------------------------------
// Projection: out[M,256] = X[M,K] @ W[K,256] + b, bf16 WMMA, fp32 accumulate.
// Block = 256 threads = 8 waves, covering the full 256 output columns:
// wave w owns N-tiles {w, w+8}, sharing one A fragment per k-step (2 WMMAs).
// The 16-row A tile is staged in LDS as bf16 once per block. grid = (M/16).
// ---------------------------------------------------------------------------
template <int K>
__global__ __launch_bounds__(256) void proj_kernel(
    const float* __restrict__ X, const v16bf* __restrict__ PW,
    const float* __restrict__ bias, float* __restrict__ out) {
  constexpr int KB  = K / 32;
  constexpr int LDK = K + 8;                 // +16B pad: keeps b128 alignment,
  __shared__ __bf16 sA[16 * LDK];            // spreads rows across banks

  const int tid   = threadIdx.x;
  const int mBase = blockIdx.x * 16;

  // cooperative load of the 16xK A tile, fp32 -> bf16
  for (int i = tid; i < 16 * K; i += 256) {
    int r = i / K, c = i % K;                // K is a power of two
    sA[r * LDK + c] = f2bf(X[(size_t)(mBase + r) * K + c]);
  }
  __syncthreads();

  const int wave = tid >> 5;
  const int lane = tid & 31;
  const int half = lane >> 4;                // 0: lanes 0-15, 1: lanes 16-31
  const int l16  = lane & 15;
  const int nt0  = wave;                     // N-tile pair for this wave
  const int nt1  = wave + 8;

  v8f c0 = {}, c1 = {};
#pragma unroll
  for (int kb = 0; kb < KB; ++kb) {
    // --- A fragment (16x32 bf16, contiguous k = 16*half + i), shared by both tiles ---
    v16bf a;
    const __bf16* arow = &sA[l16 * LDK];
    const int ka = kb * 32 + half * 16;
#pragma unroll
    for (int i = 0; i < 8; ++i) a[i]     = arow[ka + i];        // K: ka..ka+7
#pragma unroll
    for (int i = 0; i < 8; ++i) a[8 + i] = arow[ka + 8 + i];    // K: ka+8..ka+15
    // --- B fragments: pre-packed, 32 contiguous bytes per lane ---
    v16bf b0 = PW[(size_t)(nt0 * KB + kb) * 32 + lane];
    v16bf b1 = PW[(size_t)(nt1 * KB + kb) * 32 + lane];

    c0 = __builtin_amdgcn_wmma_f32_16x16x32_bf16(
        false, a, false, b0, (short)0, c0, false, false);
    c1 = __builtin_amdgcn_wmma_f32_16x16x32_bf16(
        false, a, false, b1, (short)0, c1, false, false);
  }

  const float bv0 = bias[nt0 * 16 + l16];
  const float bv1 = bias[nt1 * 16 + l16];
#pragma unroll
  for (int r = 0; r < 8; ++r) {
    const int m = mBase + half * 8 + r;      // C layout: VGPR r -> row r(+8)
    out[(size_t)m * HD + nt0 * 16 + l16] = c0[r] + bv0;
    out[(size_t)m * HD + nt1 * 16 + l16] = c1[r] + bv1;
  }
}

// ---------------------------------------------------------------------------
// Per-graph GATv2 edge phase + head maxpool. One 256-thread block per graph.
// hs staged in LDS (reused by logits and aggregation); hd streamed from L2.
// ---------------------------------------------------------------------------
__global__ __launch_bounds__(256) void gat_edge_kernel(
    const float* __restrict__ hs, const float* __restrict__ hd,
    const int* __restrict__ src, const int* __restrict__ dst,
    const float* __restrict__ attn, float* __restrict__ out) {
  __shared__ float s_hs[NPG * HD];      // 32 KB: source features of graph
  __shared__ float s_e[EPG * HEADS];    //  4 KB: logits -> exp() in place
  __shared__ float s_a[HEADS * DIM];    //  1 KB: attention vectors
  __shared__ float s_max[NPG * HEADS];
  __shared__ float s_inv[NPG * HEADS];
  __shared__ int   s_sl[EPG], s_dl[EPG];

  const int tid = threadIdx.x;
  const int g   = blockIdx.x;
  const int nb  = g * NPG, eb = g * EPG;

  for (int i = tid; i < NPG * HD; i += 256) s_hs[i] = hs[(size_t)nb * HD + i];
  for (int i = tid; i < HEADS * DIM; i += 256) s_a[i] = attn[i];
  if (tid < EPG) { s_sl[tid] = src[eb + tid] - nb; s_dl[tid] = dst[eb + tid] - nb; }
  __syncthreads();

  // ---- logits: e = a . leaky_relu(hs[src] + hd[dst]) per (edge, head) ----
  for (int t = tid; t < EPG * HEADS; t += 256) {
    const int e = t >> 2, h = t & 3;
    const float* ps = &s_hs[s_sl[e] * HD + h * DIM];
    const float* pd = &hd[(size_t)(nb + s_dl[e]) * HD + h * DIM];
    const float* pa = &s_a[h * DIM];
    float acc = 0.f;
#pragma unroll 8
    for (int k = 0; k < DIM; ++k) {
      float v = ps[k] + pd[k];
      v = v > 0.f ? v : 0.2f * v;
      acc += pa[k] * v;
    }
    s_e[t] = acc;
  }
  __syncthreads();

  // ---- segment max over incoming edges of each (node, head) ----
  if (tid < NPG * HEADS) {
    const int node = tid >> 2, h = tid & 3;
    float m = -3.402823466e38f;
    for (int e = 0; e < EPG; ++e)
      if (s_dl[e] == node) { float l = s_e[e * HEADS + h]; m = l > m ? l : m; }
    if (!(m > -3.0e38f)) m = 0.f;            // no incoming edges
    s_max[tid] = m;
  }
  __syncthreads();

  // ---- exp in place ----
  for (int t = tid; t < EPG * HEADS; t += 256) {
    const int e = t >> 2, h = t & 3;
    s_e[t] = __expf(s_e[t] - s_max[s_dl[e] * HEADS + h]);
  }
  __syncthreads();

  // ---- segment sum -> reciprocal ----
  if (tid < NPG * HEADS) {
    const int node = tid >> 2, h = tid & 3;
    float den = 0.f;
    for (int e = 0; e < EPG; ++e)
      if (s_dl[e] == node) den += s_e[e * HEADS + h];
    s_inv[tid] = 1.0f / (den == 0.f ? 1.0f : den);
  }
  __syncthreads();

  // ---- aggregation + fused head maxpool: thread = (node, 8-col chunk) ----
  {
    const int node = tid >> 3;
    const int k0   = (tid & 7) * 8;
    float acc[HEADS][8];
#pragma unroll
    for (int h = 0; h < HEADS; ++h)
#pragma unroll
      for (int k = 0; k < 8; ++k) acc[h][k] = 0.f;

    for (int e = 0; e < EPG; ++e) {
      if (s_dl[e] != node) continue;
      const int s = s_sl[e];
#pragma unroll
      for (int h = 0; h < HEADS; ++h) {
        const float al = s_e[e * HEADS + h] * s_inv[node * HEADS + h];
        const float* ps = &s_hs[s * HD + h * DIM + k0];
#pragma unroll
        for (int k = 0; k < 8; ++k) acc[h][k] += al * ps[k];
      }
    }
#pragma unroll
    for (int k = 0; k < 8; ++k) {
      float m = acc[0][k];
#pragma unroll
      for (int h = 1; h < HEADS; ++h) m = acc[h][k] > m ? acc[h][k] : m;
      out[(size_t)(nb + node) * DIM + k0 + k] = m;
    }
  }
}

// ---------------------------------------------------------------------------
// Global attention pooling: one 64-thread block per graph.
// ---------------------------------------------------------------------------
__global__ __launch_bounds__(64) void pool_kernel(
    const float* __restrict__ h, const float* __restrict__ Wg,
    const float* __restrict__ bg, float* __restrict__ out) {
  __shared__ float s_h[NPG * DIM];   // 8 KB
  __shared__ float s_g[NPG];

  const int tid = threadIdx.x;
  const int g = blockIdx.x, nb = g * NPG;

  for (int i = tid; i < NPG * DIM; i += 64) s_h[i] = h[(size_t)nb * DIM + i];
  __syncthreads();

  if (tid < NPG) {
    float acc = bg[0];
    for (int k = 0; k < DIM; ++k) acc += s_h[tid * DIM + k] * Wg[k];
    s_g[tid] = acc;
  }
  __syncthreads();

  float m = s_g[0];
  for (int n = 1; n < NPG; ++n) m = s_g[n] > m ? s_g[n] : m;
  __syncthreads();
  if (tid < NPG) s_g[tid] = __expf(s_g[tid] - m);
  __syncthreads();

  float den = 0.f;
  for (int n = 0; n < NPG; ++n) den += s_g[n];
  const float inv = 1.0f / den;

  float acc = 0.f;                       // tid == output column k
  for (int n = 0; n < NPG; ++n) acc += s_g[n] * s_h[n * DIM + tid];
  out[(size_t)g * DIM + tid] = acc * inv;
}

// ---------------------------------------------------------------------------
extern "C" void kernel_launch(void* const* d_in, const int* in_sizes, int n_in,
                              void* d_out, int out_size, void* d_ws, size_t ws_size,
                              hipStream_t stream) {
  const float* x   = (const float*)d_in[0];
  const int*   src = (const int*)  d_in[1];
  const int*   dst = (const int*)  d_in[2];
  // d_in[3] = graph_ids (implicit in contiguous layout; unused)
  const float* W1s = (const float*)d_in[4];
  const float* b1s = (const float*)d_in[5];
  const float* W1d = (const float*)d_in[6];
  const float* b1d = (const float*)d_in[7];
  const float* a1  = (const float*)d_in[8];
  const float* W2s = (const float*)d_in[9];
  const float* b2s = (const float*)d_in[10];
  const float* W2d = (const float*)d_in[11];
  const float* b2d = (const float*)d_in[12];
  const float* a2  = (const float*)d_in[13];
  const float* Wg  = (const float*)d_in[14];
  const float* bg  = (const float*)d_in[15];

  char* ws = (char*)d_ws;
  float*  hs   = (float*)(ws);                              // 64 MiB: [N, 256]
  float*  hd   = (float*)(ws + ((size_t)64  << 20));        // 64 MiB: [N, 256]
  float*  h1   = (float*)(ws + ((size_t)128 << 20));        // 16 MiB: [N, 64]
  float*  h2   = (float*)(ws + ((size_t)144 << 20));        // 16 MiB: [N, 64]
  __bf16* pW1s = (__bf16*)(ws + ((size_t)160 << 20));       // 64 KiB packed
  __bf16* pW1d = pW1s + 128 * HD;
  __bf16* pW2s = pW1d + 128 * HD;                           // 32 KiB packed
  __bf16* pW2d = pW2s + 64 * HD;

  // ---- pre-pack weights into WMMA B-fragment bf16 layout (tiny) ----
  pack_w_kernel<128><<<128, 256, 0, stream>>>(W1s, pW1s);
  pack_w_kernel<128><<<128, 256, 0, stream>>>(W1d, pW1d);
  pack_w_kernel<64><<<64, 256, 0, stream>>>(W2s, pW2s);
  pack_w_kernel<64><<<64, 256, 0, stream>>>(W2d, pW2d);

  const dim3 gproj(N_NODES / 16);

  // ---- layer 1 ----
  proj_kernel<128><<<gproj, 256, 0, stream>>>(x, (const v16bf*)pW1s, b1s, hs);
  proj_kernel<128><<<gproj, 256, 0, stream>>>(x, (const v16bf*)pW1d, b1d, hd);
  gat_edge_kernel<<<N_GRAPHS, 256, 0, stream>>>(hs, hd, src, dst, a1, h1);

  // ---- layer 2 (reuse hs/hd scratch) ----
  proj_kernel<64><<<gproj, 256, 0, stream>>>(h1, (const v16bf*)pW2s, b2s, hs);
  proj_kernel<64><<<gproj, 256, 0, stream>>>(h1, (const v16bf*)pW2d, b2d, hd);
  gat_edge_kernel<<<N_GRAPHS, 256, 0, stream>>>(hs, hd, src, dst, a2, h2);

  // ---- global attention pooling ----
  pool_kernel<<<N_GRAPHS, 64, 0, stream>>>(h2, Wg, bg, (float*)d_out);
}
